// GraphAttnTrfAggregationAndFeedForwardModule_66331474920033
// MI455X (gfx1250) — matrix operand order repaired and checked by reading, not compiled, not run on hardware
//
#include <hip/hip_runtime.h>
#include <hip/hip_bf16.h>
#include <math.h>

// ---------------------------------------------------------------------------
// Graph-attention aggregation + FFN for MI455X (gfx1250, wave32, WMMA).
// GEMMs run on v_wmma_f32_16x16x32_f16 (fp32 -> f16 staged in LDS, f32 acc).
// Block tile 128(M) x 64(N), 4 waves; wave tile 32x64 -> 8 WMMA per K-step.
// Edge softmax uses native f32 global atomics + ordered-uint atomic max.
// ---------------------------------------------------------------------------

typedef __attribute__((ext_vector_type(16))) _Float16 v16h;
typedef __attribute__((ext_vector_type(8)))  _Float16 v8h;
typedef __attribute__((ext_vector_type(8)))  float    v8f;

#define DIMC 256
#define HEADS 8
#define HD 32

// ---- ordered-uint mapping so unsigned atomicMax implements float max -------
__device__ __forceinline__ unsigned fkey(float f) {
  unsigned u = __float_as_uint(f);
  return (u & 0x80000000u) ? ~u : (u | 0x80000000u);
}
__device__ __forceinline__ float funkey(unsigned k) {
  unsigned u = (k & 0x80000000u) ? (k & 0x7fffffffu) : ~k;
  return __uint_as_float(u);
}

// ---------------------------------------------------------------------------
__global__ void k_init(unsigned* __restrict__ mkey, float* __restrict__ denom,
                       float* __restrict__ agg, int n8, int n256) {
  int i = blockIdx.x * blockDim.x + threadIdx.x;
  if (i < n8) { mkey[i] = 0u; denom[i] = 0.0f; }
  if (i < n256) agg[i] = 0.0f;
}

// ---------------------------------------------------------------------------
// Tiled WMMA GEMM: C[M,N] = A[M,K](f32) * B[K,N](f32, row-major) + bias,
// optional exact-GELU epilogue.  Block = 128 threads = 4 waves.
// Block tile 128(M) x 64(N); wave tile 32(M) x 64(N); K-step 32.
// LDS rows padded to 40 halves (80 B) -> conflict-free b128 fragment reads.
// ---------------------------------------------------------------------------
__global__ __launch_bounds__(128)
void k_gemm_wmma(const float* __restrict__ A, int lda,
                 const float* __restrict__ B, int ldb,
                 const float* __restrict__ bias,
                 float* __restrict__ C, int ldc, int ccol_off,
                 int M, int K, int N, int do_gelu) {
  __shared__ _Float16 As[128][40];  // [m][k]
  __shared__ _Float16 Bs[64][40];   // [n][k]  (B transposed into LDS)

  const int tid  = threadIdx.x;
  const int wave = tid >> 5;
  const int lane = tid & 31;
  const int m0   = blockIdx.x * 128;
  const int n0   = blockIdx.y * 64;

  v8f acc[2][4];
  #pragma unroll
  for (int i = 0; i < 2; ++i)
    #pragma unroll
    for (int j = 0; j < 4; ++j) acc[i][j] = (v8f){};

  // B staging coordinates: thread packs a K-pair for 8 consecutive n
  const int kp  = tid >> 3;          // 0..15  -> k = 2*kp, 2*kp+1
  const int bn0 = (tid & 7) * 8;     // n chunk

  for (int k0 = 0; k0 < K; k0 += 32) {
    // ---- stage A: one full 32-wide K-row per thread (128x32, fp32->f16) ----
    {
      int row = m0 + tid;
      v8h pk[4];
      if (row < M) {
        const float4* g = (const float4*)(A + (size_t)row * lda + k0);
        if (k0 + 32 < K) __builtin_prefetch(A + (size_t)row * lda + k0 + 32, 0, 3);
        #pragma unroll
        for (int j = 0; j < 4; ++j) {
          float4 t0 = g[2 * j], t1 = g[2 * j + 1];
          v8h h;
          h[0] = (_Float16)t0.x; h[1] = (_Float16)t0.y;
          h[2] = (_Float16)t0.z; h[3] = (_Float16)t0.w;
          h[4] = (_Float16)t1.x; h[5] = (_Float16)t1.y;
          h[6] = (_Float16)t1.z; h[7] = (_Float16)t1.w;
          pk[j] = h;
        }
      } else {
        #pragma unroll
        for (int j = 0; j < 4; ++j) pk[j] = (v8h){};
      }
      v8h* rowp = (v8h*)&As[tid][0];
      #pragma unroll
      for (int j = 0; j < 4; ++j) rowp[j] = pk[j];
    }
    // ---- stage B transposed with packed b32 stores: Bs[n][k] ----
    {
      const float4* g0 = (const float4*)(B + (size_t)(k0 + 2 * kp) * ldb + n0 + bn0);
      const float4* g1 = (const float4*)(B + (size_t)(k0 + 2 * kp + 1) * ldb + n0 + bn0);
      #pragma unroll
      for (int half = 0; half < 2; ++half) {
        float4 a = g0[half], b = g1[half];
        union HU { _Float16 h[2]; unsigned u; } p0, p1, p2, p3;
        p0.h[0] = (_Float16)a.x; p0.h[1] = (_Float16)b.x;
        p1.h[0] = (_Float16)a.y; p1.h[1] = (_Float16)b.y;
        p2.h[0] = (_Float16)a.z; p2.h[1] = (_Float16)b.z;
        p3.h[0] = (_Float16)a.w; p3.h[1] = (_Float16)b.w;
        int nb = bn0 + half * 4;
        *(unsigned*)&Bs[nb + 0][2 * kp] = p0.u;
        *(unsigned*)&Bs[nb + 1][2 * kp] = p1.u;
        *(unsigned*)&Bs[nb + 2][2 * kp] = p2.u;
        *(unsigned*)&Bs[nb + 3][2 * kp] = p3.u;
      }
    }
    __syncthreads();

    // ---- fragments (per ISA 7.12.2 f16 layouts) ----
    const int hi = (lane >> 4) & 1;   // lanes 16-31 hold K 8..15 and 24..31
    union U { v16h v; v8h h[2]; };

    U a[2];
    #pragma unroll
    for (int mt = 0; mt < 2; ++mt) {
      const v8h* rA = (const v8h*)&As[wave * 32 + mt * 16 + (lane & 15)][0];
      a[mt].h[0] = rA[hi];       // K = hi*8 .. +7
      a[mt].h[1] = rA[hi + 2];   // K = hi*8+16 .. +7
    }
    #pragma unroll
    for (int nt = 0; nt < 4; ++nt) {
      U b;
      const v8h* rB = (const v8h*)&Bs[nt * 16 + (lane & 15)][0];
      b.h[0] = rB[hi];           // K = hi*8 .. +7
      b.h[1] = rB[hi + 2];       // K = hi*8+16 .. +7
      #pragma unroll
      for (int mt = 0; mt < 2; ++mt) {
        acc[mt][nt] = __builtin_amdgcn_wmma_f32_16x16x32_f16(
            false, a[mt].v, false, b.v, (short)0, acc[mt][nt], false, false);
      }
    }
    __syncthreads();
  }

  // ---- epilogue: bias (+GELU) + store ----
  const int hi = (lane >> 4) & 1;
  #pragma unroll
  for (int nt = 0; nt < 4; ++nt) {
    int ncol = n0 + nt * 16 + (lane & 15);
    float bz = bias[ncol];
    #pragma unroll
    for (int mt = 0; mt < 2; ++mt) {
      v8f c = acc[mt][nt];
      #pragma unroll
      for (int r = 0; r < 8; ++r) {
        int row = m0 + wave * 32 + mt * 16 + hi * 8 + r;
        if (row < M) {
          float v = c[r] + bz;
          if (do_gelu) v = 0.5f * v * (1.0f + erff(v * 0.70710678118654752f));
          C[(size_t)row * ldc + ccol_off + ncol] = v;
        }
      }
    }
  }
}

// ---------------------------------------------------------------------------
// scores[e,h] = (k[src[e],h] . q[dst[e],h]) / sqrt(HD); segment max via
// ordered-uint atomicMax.
// ---------------------------------------------------------------------------
__global__ void k_scores(const float* __restrict__ qkv,
                         const int* __restrict__ src,
                         const int* __restrict__ dst,
                         float* __restrict__ scores,
                         unsigned* __restrict__ mkey, int E) {
  int t = blockIdx.x * blockDim.x + threadIdx.x;
  if (t >= E * HEADS) return;
  int e = t >> 3, h = t & 7;
  int s = src[e], d = dst[e];
  const float4* kk = (const float4*)(qkv + (size_t)s * 768 + h * 96 + 32);
  const float4* qq = (const float4*)(qkv + (size_t)d * 768 + h * 96);
  float acc = 0.0f;
  #pragma unroll
  for (int i = 0; i < 8; ++i) {
    float4 a = kk[i], b = qq[i];
    acc += a.x * b.x + a.y * b.y + a.z * b.z + a.w * b.w;
  }
  acc *= 0.17677669529663687f;  // 1/sqrt(32)
  scores[t] = acc;
  atomicMax(&mkey[(size_t)d * HEADS + h], fkey(acc));
}

// ---------------------------------------------------------------------------
// p = exp(s - m); denom += p; agg[dst] += p * v[src]   (native f32 atomics)
// ---------------------------------------------------------------------------
__global__ void k_edge_agg(const float* __restrict__ qkv,
                           const float* __restrict__ scores,
                           const int* __restrict__ src,
                           const int* __restrict__ dst,
                           const unsigned* __restrict__ mkey,
                           float* __restrict__ denom,
                           float* __restrict__ agg, int E) {
  int t = blockIdx.x * blockDim.x + threadIdx.x;
  if (t >= E * HEADS) return;
  int e = t >> 3, h = t & 7;
  int s = src[e], d = dst[e];
  float m = funkey(mkey[(size_t)d * HEADS + h]);
  float p = __expf(scores[t] - m);
  atomicAdd(&denom[(size_t)d * HEADS + h], p);
  const float4* vv = (const float4*)(qkv + (size_t)s * 768 + h * 96 + 64);
  float* ag = agg + (size_t)d * DIMC + h * HD;
  #pragma unroll
  for (int i = 0; i < 8; ++i) {
    float4 v4 = vv[i];
    atomicAdd(ag + 4 * i + 0, p * v4.x);
    atomicAdd(ag + 4 * i + 1, p * v4.y);
    atomicAdd(ag + 4 * i + 2, p * v4.z);
    atomicAdd(ag + 4 * i + 3, p * v4.w);
  }
}

// ---------------------------------------------------------------------------
// agg /= denom (guard empty segments); h[:, :256] = x
// ---------------------------------------------------------------------------
__global__ void k_norm_concat(const float* __restrict__ x,
                              float* __restrict__ agg,
                              const float* __restrict__ denom,
                              float* __restrict__ hbuf, int N) {
  int i = blockIdx.x * blockDim.x + threadIdx.x;
  if (i >= N * DIMC) return;
  int n = i >> 8, c = i & 255;
  float dn = denom[(size_t)n * HEADS + (c >> 5)];
  float a = agg[i];
  agg[i] = (dn > 0.0f) ? (a / dn) : 0.0f;
  hbuf[(size_t)n * 512 + c] = x[i];
}

// ---------------------------------------------------------------------------
extern "C" void kernel_launch(void* const* d_in, const int* in_sizes, int n_in,
                              void* d_out, int out_size, void* d_ws, size_t ws_size,
                              hipStream_t stream) {
  const float* x    = (const float*)d_in[0];
  const int*   src  = (const int*)d_in[1];
  const int*   dst  = (const int*)d_in[2];
  const float* Wqkv = (const float*)d_in[3];
  const float* bqkv = (const float*)d_in[4];
  const float* Wout = (const float*)d_in[5];
  const float* bout = (const float*)d_in[6];
  const float* W1   = (const float*)d_in[7];
  const float* b1   = (const float*)d_in[8];
  const float* W2   = (const float*)d_in[9];
  const float* b2   = (const float*)d_in[10];
  float* out = (float*)d_out;

  const int Nn = in_sizes[0] / DIMC;   // 50000
  const int E  = in_sizes[1];          // 800000

  // workspace carve-up
  float*    ws     = (float*)d_ws;
  float*    qkv    = ws;                                      // N*768
  float*    scores = qkv + (size_t)Nn * 768;                  // E*8
  unsigned* mkey   = (unsigned*)(scores + (size_t)E * HEADS); // N*8
  float*    denom  = (float*)(mkey + (size_t)Nn * HEADS);     // N*8
  float*    agg    = denom + (size_t)Nn * HEADS;              // N*256
  float*    hbuf   = agg + (size_t)Nn * DIMC;                 // N*512
  float*    f1     = hbuf + (size_t)Nn * 512;                 // N*256

  // 1) zero segment-max keys / denom / agg
  {
    int tot = Nn * DIMC;
    k_init<<<(tot + 255) / 256, 256, 0, stream>>>(mkey, denom, agg,
                                                  Nn * HEADS, tot);
  }
  // 2) qkv = x @ Wqkv + bqkv
  {
    dim3 g((Nn + 127) / 128, 768 / 64);
    k_gemm_wmma<<<g, 128, 0, stream>>>(x, DIMC, Wqkv, 768, bqkv,
                                       qkv, 768, 0, Nn, DIMC, 768, 0);
  }
  // 3) edge scores + segment max
  {
    int tot = E * HEADS;
    k_scores<<<(tot + 255) / 256, 256, 0, stream>>>(qkv, src, dst, scores,
                                                    mkey, E);
  }
  // 4) softmax numerators + weighted v aggregation
  {
    int tot = E * HEADS;
    k_edge_agg<<<(tot + 255) / 256, 256, 0, stream>>>(qkv, scores, src, dst,
                                                      mkey, denom, agg, E);
  }
  // 5) normalize agg; write x into h[:, :256]
  {
    int tot = Nn * DIMC;
    k_norm_concat<<<(tot + 255) / 256, 256, 0, stream>>>(x, agg, denom, hbuf, Nn);
  }
  // 6) h[:, 256:] = agg @ Wout + bout
  {
    dim3 g((Nn + 127) / 128, DIMC / 64);
    k_gemm_wmma<<<g, 128, 0, stream>>>(agg, DIMC, Wout, DIMC, bout,
                                       hbuf, 512, DIMC, Nn, DIMC, DIMC, 0);
  }
  // 7) f1 = gelu(h @ W1 + b1)
  {
    dim3 g((Nn + 127) / 128, DIMC / 64);
    k_gemm_wmma<<<g, 128, 0, stream>>>(hbuf, 512, W1, DIMC, b1,
                                       f1, DIMC, 0, Nn, 512, DIMC, 1);
  }
  // 8) out = f1 @ W2 + b2
  {
    dim3 g((Nn + 127) / 128, DIMC / 64);
    k_gemm_wmma<<<g, 128, 0, stream>>>(f1, DIMC, W2, DIMC, b2,
                                       out, DIMC, 0, Nn, DIMC, DIMC, 0);
  }
}
